// HLinearAttention_3427383902669
// MI455X (gfx1250) — matrix operand order repaired and checked by reading, not compiled, not run on hardware
//
#include <hip/hip_runtime.h>

typedef __attribute__((ext_vector_type(16))) __bf16        v16bf;
typedef __attribute__((ext_vector_type(4)))  __bf16        v4bf;
typedef __attribute__((ext_vector_type(8)))  float         v8f;
typedef __attribute__((ext_vector_type(4)))  float         v4f;
typedef __attribute__((ext_vector_type(4)))  unsigned int  v4u;

#define DIM_D  2048
#define NH     16
#define NKVH   8
#define DH     128
#define BATCH  4
#define SEQ    2048
#define TTOK   (BATCH * SEQ)      // 8192
#define CHUNK  64
#define NCHUNK (SEQ / CHUNK)      // 32
#define NQCOL  (NH * DH)          // 2048
#define NKCOL  (NKVH * DH)        // 1024
#define NTOT   ((NH + 2 * NKVH) * DH) // 4096

union V16U { v16bf bf; v4u u[2]; };

// ---------------------------------------------------------------------------
// WMMA fragment helpers (gfx1250 wave32 layouts, CDNA5 ISA §7.12.2)
// Per-lane data is contiguous: A = two 16B runs (k=kh..kh+7, 16+kh..16+kh+7),
// B (from [n][k]-stored source) = same two 16B runs as A. Strides are
// multiples of 8 elements so every run is 16B-aligned -> ds_load_b128.
// ---------------------------------------------------------------------------
__device__ __forceinline__ v8f wmma_bf16(v16bf a, v16bf b, v8f c) {
  return __builtin_amdgcn_wmma_f32_16x16x32_bf16(false, a, false, b,
                                                 (short)0, c, false, false);
}

// A (16x32 bf16), row-major source: elem(m,k) = base[m*stride + k]
__device__ __forceinline__ v16bf load_a_frag(const __bf16* base, int stride) {
  int lane = threadIdx.x & 31;
  const __bf16* p = base + (lane & 15) * stride + ((lane >> 4) << 3);
  V16U t;
  t.u[0] = *(const v4u*)p;          // k = kh .. kh+7
  t.u[1] = *(const v4u*)(p + 16);   // k = 16+kh .. 16+kh+7
  return t.bf;
}

// B (32x16 bf16) from column-stored source: elem(k,n) = base[n*stride + k]
__device__ __forceinline__ v16bf load_b_frag_ct(const __bf16* base, int stride) {
  int lane = threadIdx.x & 31;
  const __bf16* p = base + (lane & 15) * stride + ((lane >> 4) << 3);
  V16U t;
  t.u[0] = *(const v4u*)p;          // k = kh .. kh+7
  t.u[1] = *(const v4u*)(p + 16);   // k = 16+kh .. 16+kh+7
  return t.bf;
}

// ---------------------------------------------------------------------------
// Kernel 1: QKV GEMM, 128x128 tile per WG, bf16 WMMA, fused activations
// ---------------------------------------------------------------------------
__global__ __launch_bounds__(256) void qkv_gemm_kernel(
    const float* __restrict__ X, const float* __restrict__ W,
    const float* __restrict__ bias, float* __restrict__ qws,
    float* __restrict__ kws, float* __restrict__ gws,
    unsigned short* __restrict__ vws_u) {
  __shared__ __align__(16) __bf16 As[128 * 40];    // [m][k]
  __shared__ __align__(16) __bf16 BsT[128 * 40];   // [n][k]  (transposed)
  __bf16* vws = reinterpret_cast<__bf16*>(vws_u);

  int tid = threadIdx.x;
  int bx = blockIdx.x, by = blockIdx.y;
  int wv = tid >> 5, lane = tid & 31;
  int wm = wv >> 2, wn = wv & 3;    // 2x4 wave grid over the 128x128 tile

  v8f acc[4][2];
#pragma unroll
  for (int r = 0; r < 4; ++r)
#pragma unroll
    for (int c = 0; c < 2; ++c) acc[r][c] = (v8f){};

  for (int kt = 0; kt < DIM_D / 32; ++kt) {
    // prefetch next stage into L2 (global_prefetch_b8)
    if (kt + 1 < DIM_D / 32) {
      __builtin_prefetch(&X[(size_t)(by * 128 + (tid & 127)) * DIM_D + (kt + 1) * 32], 0, 1);
      __builtin_prefetch(&W[(size_t)((kt + 1) * 32 + (tid & 31)) * NTOT + bx * 128], 0, 1);
    }
    // A tile 128x32: float4 loads, packed v4bf stores
#pragma unroll
    for (int e = 0; e < 4; ++e) {
      int lin = tid + e * 256;               // 1024 groups of 4
      int r = lin >> 3, k4 = (lin & 7) * 4;
      v4f xv = *(const v4f*)&X[(size_t)(by * 128 + r) * DIM_D + kt * 32 + k4];
      v4bf xb;
#pragma unroll
      for (int j = 0; j < 4; ++j) xb[j] = (__bf16)xv[j];
      *(v4bf*)&As[r * 40 + k4] = xb;
    }
    // B tile 32x128 stored [n][k]: 4 coalesced b32 loads -> one v4bf store
#pragma unroll
    for (int e = 0; e < 4; ++e) {
      int lin = tid + e * 256;
      int nn = lin & 127, kr4 = (lin >> 7) * 4;
      v4bf wb;
#pragma unroll
      for (int j = 0; j < 4; ++j)
        wb[j] = (__bf16)W[(size_t)(kt * 32 + kr4 + j) * NTOT + bx * 128 + nn];
      *(v4bf*)&BsT[nn * 40 + kr4] = wb;
    }
    __syncthreads();

    v16bf afr[4], bfr[2];
#pragma unroll
    for (int rt = 0; rt < 4; ++rt)
      afr[rt] = load_a_frag(As + (wm * 64 + rt * 16) * 40, 40);
#pragma unroll
    for (int ct = 0; ct < 2; ++ct)
      bfr[ct] = load_b_frag_ct(BsT + (wn * 32 + ct * 16) * 40, 40);
#pragma unroll
    for (int rt = 0; rt < 4; ++rt)
#pragma unroll
      for (int ct = 0; ct < 2; ++ct)
        acc[rt][ct] = wmma_bf16(afr[rt], bfr[ct], acc[rt][ct]);
    __syncthreads();
  }

  // epilogue: bias + per-column activation (tile is uniformly q, k, or v)
  int n = lane & 15, hi = lane >> 4;
#pragma unroll
  for (int rt = 0; rt < 4; ++rt) {
#pragma unroll
    for (int ct = 0; ct < 2; ++ct) {
      int col = bx * 128 + wn * 32 + ct * 16 + n;
      float bv = bias[col];
#pragma unroll
      for (int m = 0; m < 8; ++m) {
        int row = by * 128 + wm * 64 + rt * 16 + m + 8 * hi;
        float val = acc[rt][ct][m] + bv;
        if (bx < 16) {                           // q: relu * DH^-0.5
          qws[(size_t)row * NQCOL + col] = fmaxf(val, 0.f) * 0.08838834764831843f;
        } else if (bx < 24) {                    // k: min(sigmoid, .95); g
          float s = 1.f / (1.f + expf(-val));
          s = fminf(s, 0.95f);
          int cc = col - NQCOL;
          kws[(size_t)row * NKCOL + cc] = s;
          gws[(size_t)row * NKCOL + cc] = log1pf(-s);
        } else {                                 // v: identity, bf16
          vws[(size_t)row * NKCOL + (col - NQCOL - NKCOL)] = (__bf16)val;
        }
      }
    }
  }
}

// ---------------------------------------------------------------------------
// Kernel 2: chunked GLA scan. One WG per (b,h). 256 threads = 8 waves.
// Dynamic LDS layout (16B-aligned slabs), total 267264 B < 320 KB WGP LDS:
//   S    : 128*130 f32  = 66560  (recurrent state, f32 master)
//   Bc   : 64*128  f32  = 32768  (gates -> cumulative decay)
//   SbT  : 128*136 bf16 = 34816  (bf16 transposed mirror of S, [dv][dk])
//   qS   : 64*136  bf16 = 17408  (q * exp(Bc); also jt=0 rebased q)
//   qe   : 3*64*136bf16 = 52224  (q rebased at B*_1..3)
//   ke   : 64*136  bf16 = 17408  ([j][d]; contiguous as B via [n][k] reads)
//   kgT  : 128*72  bf16 = 18432  ([dk][tok])
//   vvT  : 128*72  bf16 = 18432  ([dv][tok])
//   Als  : 64*72   bf16 =  9216  (intra-chunk A)
// ---------------------------------------------------------------------------
#define GLA_SMEM 267264

__global__ __launch_bounds__(256) void gla_scan_kernel(
    const float* __restrict__ qws, const float* __restrict__ kws,
    const float* __restrict__ gws, const unsigned short* __restrict__ vws_u,
    const float* __restrict__ kv_cache, const int* __restrict__ state_indices,
    float* __restrict__ out) {
  extern __shared__ char smem_raw[];
  float*  S   = (float*)smem_raw;                 // stride 130
  float*  Bc  = S + 128 * 130;                    // stride 128
  __bf16* SbT = (__bf16*)(Bc + 64 * 128);         // stride 136, [dv][dk]
  __bf16* qS  = SbT + 128 * 136;                  // stride 136
  __bf16* qe  = qS + 64 * 136;                    // 3 slabs, stride 136
  __bf16* ke  = qe + 3 * 64 * 136;                // stride 136, [j][d]
  __bf16* kgT = ke + 64 * 136;                    // stride 72,  [dk][tok]
  __bf16* vvT = kgT + 128 * 72;                   // stride 72,  [dv][tok]
  __bf16* Als = vvT + 128 * 72;                   // stride 72
  const __bf16* vws = reinterpret_cast<const __bf16*>(vws_u);

  int tid = threadIdx.x;
  int b = blockIdx.x >> 4, h = blockIdx.x & 15;
  int kvh = h >> 1;
  int wv = tid >> 5, lane = tid & 31;
  int n = lane & 15, hi = lane >> 4;

  // ---- load initial state S0 = kv_cache[slot, h]; keep f32 + bf16^T ----
  {
    int slot = state_indices[b];
    const float* s0 = kv_cache + ((size_t)slot * NH + h) * DH * DH;
#pragma unroll 4
    for (int e = 0; e < 16; ++e) {
      int lin = tid + e * 256;                    // 4096 groups of 4
      int dk = lin >> 5, dv4 = (lin & 31) * 4;
      v4f vsl = *(const v4f*)&s0[dk * DH + dv4];
      *(v4f*)&S[dk * 130 + dv4] = vsl;
#pragma unroll
      for (int j = 0; j < 4; ++j) SbT[(dv4 + j) * 136 + dk] = (__bf16)vsl[j];
    }
  }
  __syncthreads();

  for (int ch = 0; ch < NCHUNK; ++ch) {
    int t0 = b * SEQ + ch * CHUNK;

    // ---- stage gates via gfx1250 async copy (global -> LDS, no VGPRs) ----
    {
      unsigned bc_base = (unsigned)(size_t)(void*)Bc;   // LDS byte offset
      const float* gsrc = gws + (size_t)t0 * NKCOL + kvh * DH;
#pragma unroll
      for (int e = 0; e < 8; ++e) {
        int lin = (tid + e * 256) * 4;                  // 4 floats per op
        int i = lin >> 7, d = lin & 127;
        unsigned lds_b = bc_base + (unsigned)lin * 4u;
        unsigned gb = (unsigned)(i * NKCOL + d) * 4u;
        asm volatile("global_load_async_to_lds_b128 %0, %1, %2 offset:0"
                     :: "v"(lds_b), "v"(gb), "s"(gsrc) : "memory");
      }
      asm volatile("s_wait_asynccnt 0" ::: "memory");
    }
    __syncthreads();

    // ---- cumsum along tokens (one column of 64 per thread) ----
    if (tid < 128) {
      float r = 0.f;
      for (int i = 0; i < 64; ++i) {
        r += Bc[i * 128 + tid];
        Bc[i * 128 + tid] = r;
      }
    }
    __syncthreads();

    // ---- build bf16 operand tiles (4 consecutive d per thread) ----
#pragma unroll 2
    for (int e = 0; e < 8; ++e) {
      int lin = tid + e * 256;                    // 2048 groups of 4
      int i = lin >> 5, d4 = (lin & 31) * 4;
      v4f bc = *(const v4f*)&Bc[i * 128 + d4];
      v4f b1 = *(const v4f*)&Bc[15 * 128 + d4];
      v4f b2 = *(const v4f*)&Bc[31 * 128 + d4];
      v4f b3 = *(const v4f*)&Bc[47 * 128 + d4];
      v4f bl = *(const v4f*)&Bc[63 * 128 + d4];
      v4f qv = *(const v4f*)&qws[(size_t)(t0 + i) * NQCOL + h * DH + d4];
      v4f kv = *(const v4f*)&kws[(size_t)(t0 + i) * NKCOL + kvh * DH + d4];
      v4bf vv4 = *(const v4bf*)&vws[(size_t)(t0 + i) * NKCOL + kvh * DH + d4];
      v4f zero = {0.f, 0.f, 0.f, 0.f};
      v4f bstk = (i < 16) ? zero : (i < 32) ? b1 : (i < 48) ? b2 : b3;
      v4bf o0, o1, o2, o3, oke;
#pragma unroll
      for (int j = 0; j < 4; ++j) {
        o0[j]  = (__bf16)(qv[j] * expf(bc[j]));
        o1[j]  = (__bf16)(qv[j] * expf(bc[j] - b1[j]));
        o2[j]  = (__bf16)(qv[j] * expf(bc[j] - b2[j]));
        o3[j]  = (__bf16)(qv[j] * expf(bc[j] - b3[j]));
        oke[j] = (__bf16)(kv[j] * expf(bstk[j] - bc[j]));
        kgT[(d4 + j) * 72 + i] = (__bf16)(kv[j] * expf(bl[j] - bc[j]));
        vvT[(d4 + j) * 72 + i] = vv4[j];
      }
      *(v4bf*)&qS[i * 136 + d4]                = o0;   // jt=0
      *(v4bf*)&qe[0 * 64 * 136 + i * 136 + d4] = o1;
      *(v4bf*)&qe[1 * 64 * 136 + i * 136 + d4] = o2;
      *(v4bf*)&qe[2 * 64 * 136 + i * 136 + d4] = o3;
      *(v4bf*)&ke[i * 136 + d4]                = oke;
    }
    __syncthreads();

    // ---- intra-chunk A matrix: 16 tiles (it,jt), 2 per wave ----
#pragma unroll
    for (int rep = 0; rep < 2; ++rep) {
      int tt = wv + rep * 8;
      int it = tt >> 2, jt = tt & 3;
      if (jt > it) {      // strictly upper tile: fully masked -> zero
#pragma unroll
        for (int m = 0; m < 8; ++m)
          Als[(it * 16 + m + 8 * hi) * 72 + jt * 16 + n] = (__bf16)0.0f;
      } else {
        const __bf16* qsrc = (jt == 0) ? qS : (qe + (jt - 1) * 64 * 136);
        v8f acc = (v8f){};
#pragma unroll
        for (int kk = 0; kk < 4; ++kk) {
          v16bf a  = load_a_frag(qsrc + (it * 16) * 136 + kk * 32, 136);
          v16bf bf = load_b_frag_ct(ke + (jt * 16) * 136 + kk * 32, 136);
          acc = wmma_bf16(a, bf, acc);
        }
#pragma unroll
        for (int m = 0; m < 8; ++m) {
          int iloc = m + 8 * hi;
          float val = acc[m];
          if (it == jt && iloc < n) val = 0.f;   // causal mask on diagonal
          Als[(it * 16 + iloc) * 72 + jt * 16 + n] = (__bf16)val;
        }
      }
    }
    __syncthreads();

    // ---- o = qS @ S + A @ v ; wave: row-block rb, column half ----
    {
      int rb = wv >> 1, half = wv & 1;
#pragma unroll
      for (int ct = 0; ct < 4; ++ct) {
        int n0 = half * 64 + ct * 16;
        v8f acc = (v8f){};
#pragma unroll
        for (int kk = 0; kk < 4; ++kk) {         // inter-chunk: K = 128
          v16bf a  = load_a_frag(qS + (rb * 16) * 136 + kk * 32, 136);
          v16bf bf = load_b_frag_ct(SbT + n0 * 136 + kk * 32, 136);
          acc = wmma_bf16(a, bf, acc);
        }
#pragma unroll
        for (int kk = 0; kk < 2; ++kk) {         // intra-chunk: K = 64
          v16bf a  = load_a_frag(Als + (rb * 16) * 72 + kk * 32, 72);
          v16bf bf = load_b_frag_ct(vvT + n0 * 72 + kk * 32, 72);
          acc = wmma_bf16(a, bf, acc);
        }
#pragma unroll
        for (int m = 0; m < 8; ++m) {
          int row = t0 + rb * 16 + m + 8 * hi;
          out[(size_t)row * NQCOL + h * DH + n0 + n] = acc[m];
        }
      }
    }
    __syncthreads();

    // ---- state update: S = S * exp(b_last)[k] + kg^T @ v (f32 + bf16^T) ----
    {
#pragma unroll
      for (int vn = 0; vn < 8; ++vn) {
        v8f acc;
#pragma unroll
        for (int m = 0; m < 8; ++m) {
          int krow = wv * 16 + m + 8 * hi;
          acc[m] = S[krow * 130 + vn * 16 + n] * expf(Bc[63 * 128 + krow]);
        }
#pragma unroll
        for (int kk = 0; kk < 2; ++kk) {
          v16bf a  = load_a_frag(kgT + (wv * 16) * 72 + kk * 32, 72);
          v16bf bf = load_b_frag_ct(vvT + (vn * 16) * 72 + kk * 32, 72);
          acc = wmma_bf16(a, bf, acc);
        }
#pragma unroll
        for (int m = 0; m < 8; ++m) {
          int krow = wv * 16 + m + 8 * hi;
          S[krow * 130 + vn * 16 + n] = acc[m];
          SbT[(vn * 16 + n) * 136 + krow] = (__bf16)acc[m];
        }
      }
    }
    __syncthreads();
  }
}

// ---------------------------------------------------------------------------
extern "C" void kernel_launch(void* const* d_in, const int* in_sizes, int n_in,
                              void* d_out, int out_size, void* d_ws, size_t ws_size,
                              hipStream_t stream) {
  const float* hidden  = (const float*)d_in[0];
  const float* W_qkv   = (const float*)d_in[1];
  const float* b_qkv   = (const float*)d_in[2];
  const float* kvcache = (const float*)d_in[3];
  const int*   sidx    = (const int*)d_in[4];
  float* out = (float*)d_out;

  // workspace carve-up
  float* qws = (float*)d_ws;                                   // T * 2048 f32
  float* kws = qws + (size_t)TTOK * NQCOL;                     // T * 1024 f32
  float* gws = kws + (size_t)TTOK * NKCOL;                     // T * 1024 f32
  unsigned short* vws = (unsigned short*)(gws + (size_t)TTOK * NKCOL); // T*1024 bf16

  (void)hipFuncSetAttribute(reinterpret_cast<const void*>(gla_scan_kernel),
                            hipFuncAttributeMaxDynamicSharedMemorySize, GLA_SMEM);

  dim3 g1(NTOT / 128, TTOK / 128, 1);
  qkv_gemm_kernel<<<g1, 256, 0, stream>>>(hidden, W_qkv, b_qkv, qws, kws, gws, vws);

  gla_scan_kernel<<<dim3(BATCH * NH), 256, GLA_SMEM, stream>>>(
      qws, kws, gws, vws, kvcache, sidx, out);
}
